// Attention_51436528337044
// MI455X (gfx1250) — matrix operand order repaired and checked by reading, not compiled, not run on hardware
//
#include <hip/hip_runtime.h>
#include <hip/hip_bf16.h>

// ---- sizes (compile-time, match reference) ----
#define S_SEQ 4096
#define D_MOD 1024
#define H_HEADS 16
#define DK 64
#define LN_EPS 1e-5f

typedef __attribute__((ext_vector_type(16))) __bf16 v16bf;
typedef __attribute__((ext_vector_type(8)))  float  v8f;
typedef int v4i __attribute__((vector_size(4 * sizeof(int))));

// ---------------------------------------------------------------------------
// CDNA5 async global->LDS path (ASYNCcnt-tracked), guarded so the file still
// compiles if the toolchain lacks the builtin.
// Builtin prototype (from hipcc diagnostic): arg0 is global ptr to int32x4.
// ---------------------------------------------------------------------------
#if defined(__has_builtin)
#if __has_builtin(__builtin_amdgcn_global_load_async_to_lds_b128)
#define ASYNC_LDS 1
#endif
#endif

#ifdef ASYNC_LDS
__device__ inline void async_copy_b128(const void* gsrc, void* ldst) {
  // generic->as1: same 64-bit value; generic LDS addr low 32 bits == LDS offset
  __builtin_amdgcn_global_load_async_to_lds_b128(
      (__attribute__((address_space(1))) v4i*)(unsigned long long)gsrc,
      (__attribute__((address_space(3))) v4i*)(unsigned int)(unsigned long long)ldst,
      0, 0);
}
__device__ inline void wait_async0() {
#if __has_builtin(__builtin_amdgcn_s_wait_asynccnt)
  __builtin_amdgcn_s_wait_asynccnt(0);
#else
  asm volatile("s_wait_asynccnt 0" ::: "memory");
#endif
}
#endif

// ---------------------------------------------------------------------------
// Fragment loaders (CDNA5 WMMA wave32 layouts, CDNA5 ISA §7.12.2)
// 16-bit A 16x32: lane L holds row M=L%16; elem j: K=(L/16)*16+j (contiguous; this code's order, not the ISA table's)
// 16-bit B 32x16: lane L holds col N=L%16; elem e: K=(L/16)*16+e (contiguous; this code's order, not the ISA table's)
// 32-bit C/D 16x16: VGPR r, lane L -> (M = r + 8*(L/16), N = L%16)
// ---------------------------------------------------------------------------
__device__ inline v16bf load_b_frag(const __bf16* p) {
  v16bf v;
#pragma unroll
  for (int i = 0; i < 16; ++i) v[i] = p[i];
  return v;
}

__device__ inline v16bf load_a_frag_bf(const __bf16* row, int hi) {
  v16bf v;
#pragma unroll
  for (int j = 0; j < 8; ++j) {
    v[j]     = row[2 * hi + j];
    v[j + 8] = row[2 * hi + 8 + j];
  }
  return v;
}

__device__ inline v16bf load_a_frag_f32(const float* row, int hi) {
  v16bf v;
#pragma unroll
  for (int j = 0; j < 8; ++j) {
    v[j]     = (__bf16)row[2 * hi + j];
    v[j + 8] = (__bf16)row[2 * hi + 8 + j];
  }
  return v;
}

__device__ inline v8f wmma_bf16(v16bf a, v16bf b, v8f c) {
  return __builtin_amdgcn_wmma_f32_16x16x32_bf16(
      false, a, false, b, (short)0, c, false, false);
}

__device__ inline v8f v8f_zero() {
  v8f c;
#pragma unroll
  for (int r = 0; r < 8; ++r) c[r] = 0.0f;
  return c;
}

// ---------------------------------------------------------------------------
// Kernel 1: per-head QKV projection.  q = (x_h @ W^T + b) [Q scaled 1/8]
// Outputs (bf16): Q[h][s][e], K[h][s][e], Vt[h][e][s]
// grid (S/128, H), block 256 (8 waves x 16 rows)
// ---------------------------------------------------------------------------
__global__ __launch_bounds__(256) void qkv_proj_kernel(
    const float* __restrict__ x,
    const float* __restrict__ Wq, const float* __restrict__ bq,
    const float* __restrict__ Wk, const float* __restrict__ bk,
    const float* __restrict__ Wv, const float* __restrict__ bv,
    __bf16* __restrict__ Qg, __bf16* __restrict__ Kg, __bf16* __restrict__ Vtg)
{
  __shared__ __bf16 Wl[3][DK * DK];
  __shared__ float  bl[3][DK];

  const int h     = blockIdx.y;
  const int sBase = blockIdx.x * 128;
  const int tid   = threadIdx.x;

  const float* Ws[3] = {Wq + h * DK * DK, Wk + h * DK * DK, Wv + h * DK * DK};
  const float* bs[3] = {bq + h * DK,      bk + h * DK,      bv + h * DK};
#pragma unroll
  for (int m = 0; m < 3; ++m) {
    for (int i = tid; i < DK * DK; i += 256) Wl[m][i] = (__bf16)Ws[m][i];
    if (tid < DK) bl[m][tid] = bs[m][tid];
  }
  __syncthreads();

  const int wave = tid >> 5, lane = tid & 31;
  const int hi   = (lane >> 4) * 8;
  const int ksub = (lane >> 4) * 16;
  const int nlo  = lane & 15;

  const float* xrow = x + (size_t)(sBase + wave * 16 + nlo) * D_MOD + h * DK;
  v16bf a0 = load_a_frag_f32(xrow, hi);
  v16bf a1 = load_a_frag_f32(xrow + 32, hi);

#pragma unroll
  for (int m = 0; m < 3; ++m) {
    const float scale = (m == 0) ? 0.125f : 1.0f;  // 1/sqrt(64) folded into Q
#pragma unroll
    for (int nt = 0; nt < 4; ++nt) {
      const int ncol = nt * 16 + nlo;
      // C starts at inline 0 (WMMA SRC2 inline constant); bias added at store.
      v8f c = v8f_zero();
      v16bf b0 = load_b_frag(&Wl[m][ncol * DK + 0  + ksub]);
      c = wmma_bf16(a0, b0, c);
      v16bf b1 = load_b_frag(&Wl[m][ncol * DK + 32 + ksub]);
      c = wmma_bf16(a1, b1, c);
      const float bias = bl[m][ncol];
#pragma unroll
      for (int r = 0; r < 8; ++r) {
        const int srow = sBase + wave * 16 + r + 8 * (lane >> 4);
        const float val = (c[r] + bias) * scale;
        if (m == 0)      Qg[((size_t)h * S_SEQ + srow) * DK + ncol] = (__bf16)val;
        else if (m == 1) Kg[((size_t)h * S_SEQ + srow) * DK + ncol] = (__bf16)val;
        else             Vtg[(size_t)h * DK * S_SEQ + (size_t)ncol * S_SEQ + srow] = (__bf16)val;
      }
    }
  }
}

// ---------------------------------------------------------------------------
// Kernel 2: flash attention per head.  O[s][h*64+e] (f32)
// grid (S/128, H), block 256 (8 waves, 16 rows/wave)
// Double-buffered async global->LDS pipeline when available.
// ---------------------------------------------------------------------------
__global__ __launch_bounds__(256) void flash_attn_kernel(
    const __bf16* __restrict__ Qg, const __bf16* __restrict__ Kg,
    const __bf16* __restrict__ Vtg, float* __restrict__ Og)
{
  __shared__ __bf16 Kl[2][64 * 64];   // [buf][t_local][e]
  __shared__ __bf16 Vl[2][64 * 64];   // [buf][e][t_local]
  __shared__ __bf16 Pl[8][16 * 64];   // per-wave P scratch (C->A relayout)

  const int h     = blockIdx.y;
  const int sBase = blockIdx.x * 128;
  const int tid   = threadIdx.x, wave = tid >> 5, lane = tid & 31;
  const int hi    = (lane >> 4) * 8;
  const int ksub  = (lane >> 4) * 16;
  const int nlo   = lane & 15;

  const __bf16* Kh = Kg  + (size_t)h * S_SEQ * DK;
  const __bf16* Vh = Vtg + (size_t)h * DK * S_SEQ;

  const __bf16* qrow = Qg + ((size_t)h * S_SEQ + sBase + wave * 16 + nlo) * DK;
  v16bf aq0 = load_a_frag_bf(qrow, hi);
  v16bf aq1 = load_a_frag_bf(qrow + 32, hi);

  v8f oacc[4];
  float mrow[8], lsum[8];
#pragma unroll
  for (int nt = 0; nt < 4; ++nt) oacc[nt] = v8f_zero();
#pragma unroll
  for (int r = 0; r < 8; ++r) { mrow[r] = -1e30f; lsum[r] = 0.0f; }

#ifdef ASYNC_LDS
  // --- issue one 64-col K/V tile into LDS buffer `buf` (each lane 16B x4) ---
  auto issue_tile = [&](int t0, int buf) {
    const __bf16* ks = Kh + (size_t)t0 * DK;   // contiguous 8KB
#pragma unroll
    for (int i = tid; i < 512; i += 256)
      async_copy_b128(ks + i * 8, &Kl[buf][i * 8]);
#pragma unroll
    for (int i = tid; i < 512; i += 256) {
      const int e = i >> 3, c = i & 7;
      async_copy_b128(Vh + (size_t)e * S_SEQ + t0 + c * 8, &Vl[buf][i * 8]);
    }
  };
  issue_tile(0, 0);
#endif

  const int NT = S_SEQ / 64;
  for (int tile = 0; tile < NT; ++tile) {
    const int t0  = tile * 64;
    const int buf = tile & 1;

#ifdef ASYNC_LDS
    wait_async0();      // loads for buf (issued last iter / prologue) done
    __syncthreads();    // visible to all waves; prior reads of buf^1 retired
    if (tile + 1 < NT) issue_tile(t0 + 64, buf ^ 1);   // overlap with compute
#else
    __syncthreads();
    {
      const uint4* src = (const uint4*)(Kh + (size_t)t0 * DK);
      uint4* dst = (uint4*)Kl[0];
#pragma unroll
      for (int i = tid; i < 512; i += 256) dst[i] = src[i];
#pragma unroll
      for (int i = tid; i < 512; i += 256) {
        const int e = i >> 3, c = i & 7;
        ((uint4*)Vl[0])[i] = *(const uint4*)(Vh + (size_t)e * S_SEQ + t0 + c * 8);
      }
      if (t0 + 64 < S_SEQ)
        __builtin_prefetch(Kh + (size_t)(t0 + 64) * DK, 0, 0);
    }
    __syncthreads();
#endif

#ifdef ASYNC_LDS
    const __bf16* Kt = Kl[buf];
    const __bf16* Vt = Vl[buf];
#else
    const __bf16* Kt = Kl[0];
    const __bf16* Vt = Vl[0];
#endif

    // --- scores: S = Q @ K^T (Q pre-scaled). B(K=e,N=t)=Ktile[N][K] ---
    v8f sacc[4];
#pragma unroll
    for (int nt = 0; nt < 4; ++nt) {
      v8f c = v8f_zero();
      const int brow = (nt * 16 + nlo) * 64;
      v16bf b0 = load_b_frag(&Kt[brow + 0  + ksub]);
      c = wmma_bf16(aq0, b0, c);
      v16bf b1 = load_b_frag(&Kt[brow + 32 + ksub]);
      c = wmma_bf16(aq1, b1, c);
      sacc[nt] = c;
    }

    // --- online softmax (row M = r + 8*(lane>>4), spread over 16 lanes) ---
#pragma unroll
    for (int r = 0; r < 8; ++r) {
      float t = fmaxf(fmaxf(sacc[0][r], sacc[1][r]), fmaxf(sacc[2][r], sacc[3][r]));
#pragma unroll
      for (int mk = 1; mk < 16; mk <<= 1) t = fmaxf(t, __shfl_xor(t, mk, 32));
      const float mnew  = fmaxf(mrow[r], t);
      const float alpha = __expf(mrow[r] - mnew);
      float rs = 0.0f;
#pragma unroll
      for (int nt = 0; nt < 4; ++nt) {
        const float p = __expf(sacc[nt][r] - mnew);
        sacc[nt][r] = p;
        rs += p;
      }
#pragma unroll
      for (int mk = 1; mk < 16; mk <<= 1) rs += __shfl_xor(rs, mk, 32);
      lsum[r] = lsum[r] * alpha + rs;
#pragma unroll
      for (int nt = 0; nt < 4; ++nt) oacc[nt][r] *= alpha;
      mrow[r] = mnew;
    }

    // --- C-layout -> A-layout relayout of P through per-wave LDS ---
    __bf16* Pw = Pl[wave];
#pragma unroll
    for (int nt = 0; nt < 4; ++nt)
#pragma unroll
      for (int r = 0; r < 8; ++r)
        Pw[(r + 8 * (lane >> 4)) * 64 + nt * 16 + nlo] = (__bf16)sacc[nt][r];

    const __bf16* prow = Pw + nlo * 64;
    v16bf ap0 = load_a_frag_bf(prow, hi);
    v16bf ap1 = load_a_frag_bf(prow + 32, hi);

    // --- O += P @ V.  B(K=t,N=e) = Vt_tile[N][K], contiguous in K ---
#pragma unroll
    for (int nt = 0; nt < 4; ++nt) {
      const int brow = (nt * 16 + nlo) * 64;
      v16bf b0 = load_b_frag(&Vt[brow + 0  + ksub]);
      oacc[nt] = wmma_bf16(ap0, b0, oacc[nt]);
      v16bf b1 = load_b_frag(&Vt[brow + 32 + ksub]);
      oacc[nt] = wmma_bf16(ap1, b1, oacc[nt]);
    }
  }

  // --- finalize: O /= lsum, write f32 into [S][D] at head offset ---
  float inv[8];
#pragma unroll
  for (int r = 0; r < 8; ++r) inv[r] = 1.0f / lsum[r];
#pragma unroll
  for (int nt = 0; nt < 4; ++nt)
#pragma unroll
    for (int r = 0; r < 8; ++r) {
      const int srow = sBase + wave * 16 + r + 8 * (lane >> 4);
      Og[(size_t)srow * D_MOD + h * DK + nt * 16 + nlo] = oacc[nt][r] * inv[r];
    }
}

// ---------------------------------------------------------------------------
// Kernel 3: LayerNorm + residual.  out = x + LN(A)*gamma + beta
// ---------------------------------------------------------------------------
__global__ __launch_bounds__(256) void ln_res_kernel(
    const float* __restrict__ x, const float* __restrict__ A,
    const float* __restrict__ gamma, const float* __restrict__ beta,
    float* __restrict__ out)
{
  __shared__ float red[2][8];
  const int row = blockIdx.x;
  const int tid = threadIdx.x, wave = tid >> 5, lane = tid & 31;
  const float* a = A + (size_t)row * D_MOD;

  float s = 0.0f, s2 = 0.0f;
  for (int i = tid; i < D_MOD; i += 256) {
    const float v = a[i];
    s += v; s2 += v * v;
  }
#pragma unroll
  for (int mk = 1; mk < 32; mk <<= 1) {
    s  += __shfl_xor(s,  mk, 32);
    s2 += __shfl_xor(s2, mk, 32);
  }
  if (lane == 0) { red[0][wave] = s; red[1][wave] = s2; }
  __syncthreads();
  float ts = 0.0f, ts2 = 0.0f;
#pragma unroll
  for (int w = 0; w < 8; ++w) { ts += red[0][w]; ts2 += red[1][w]; }
  const float mu   = ts * (1.0f / D_MOD);
  const float var  = ts2 * (1.0f / D_MOD) - mu * mu;
  const float rstd = rsqrtf(var + LN_EPS);

  for (int i = tid; i < D_MOD; i += 256) {
    const float an = (a[i] - mu) * rstd * gamma[i] + beta[i];
    out[(size_t)row * D_MOD + i] = x[(size_t)row * D_MOD + i] + an;
  }
}

// ---------------------------------------------------------------------------
extern "C" void kernel_launch(void* const* d_in, const int* in_sizes, int n_in,
                              void* d_out, int out_size, void* d_ws, size_t ws_size,
                              hipStream_t stream) {
  (void)in_sizes; (void)n_in; (void)out_size; (void)ws_size;
  const float* x     = (const float*)d_in[0];
  const float* Wq    = (const float*)d_in[1];
  const float* bq    = (const float*)d_in[2];
  const float* Wk    = (const float*)d_in[3];
  const float* bk    = (const float*)d_in[4];
  const float* Wv    = (const float*)d_in[5];
  const float* bv    = (const float*)d_in[6];
  const float* gamma = (const float*)d_in[7];
  const float* beta  = (const float*)d_in[8];

  char* ws = (char*)d_ws;
  const size_t QKV_BYTES = (size_t)H_HEADS * S_SEQ * DK * sizeof(__bf16);  // 8 MB each
  __bf16* Qg  = (__bf16*)(ws);
  __bf16* Kg  = (__bf16*)(ws + QKV_BYTES);
  __bf16* Vtg = (__bf16*)(ws + 2 * QKV_BYTES);
  float*  Og  = (float*)(ws + 3 * QKV_BYTES);                              // 16 MB

  dim3 grid(S_SEQ / 128, H_HEADS);
  qkv_proj_kernel<<<grid, 256, 0, stream>>>(x, Wq, bq, Wk, bk, Wv, bv, Qg, Kg, Vtg);
  flash_attn_kernel<<<grid, 256, 0, stream>>>(Qg, Kg, Vtg, Og);
  ln_res_kernel<<<S_SEQ, 256, 0, stream>>>(x, Og, gamma, beta, (float*)d_out);
}
